// LSTMAutoEncoder_65223373357510
// MI455X (gfx1250) — matrix operand order repaired and checked by reading, not compiled, not run on hardware
//
#include <hip/hip_runtime.h>
#include <math.h>

// ---------------------------------------------------------------------------
// LSTM autoencoder for MI455X (gfx1250, wave32, WMMA bf16 16x16x32).
//
// B=64, T=1024, OBS=128, HID=512, LAT=256.
// Sequential recurrence dominates -> persistent 8-WG kernel with device-scope
// software barrier; per-step gate GEMM fused over [x_t | h_prev] with four
// WMMA accumulators (i,f,g,o) per 16x16 hidden tile. K loops are split per
// source segment and fully unrolled (compile-time trip counts) so B-fragment
// loads pipeline across iterations instead of stalling each wmma.
// ---------------------------------------------------------------------------

typedef __attribute__((ext_vector_type(16))) __bf16 v16bf;
typedef __attribute__((ext_vector_type(8)))  float  v8f;

#define B_    64
#define T_    1024
#define OBS_  128
#define HID_  512
#define LAT_  256

#define NWG          8
#define THREADS      256
#define WAVES_PER_WG (THREADS / 32)
#define TOTAL_WAVES  (NWG * WAVES_PER_WG)   // 64 waves for the sequential kernel

// ----------------------------- WMMA helpers --------------------------------

// A fragment, 16x32 bf16 (MxK), row-major source with leading dim `ld`.
// Layout (matches B): lanes 0-15 row M=lane, K=0..15;
// lanes 16-31 row M=lane-16, K=16..31.
__device__ __forceinline__ v16bf load_a_frag(const __bf16* __restrict__ src, int ld) {
  const int lane = threadIdx.x & 31;
  const int m    = lane & 15;
  const int kh   = (lane >> 4) << 4;  // 0 or 16
  const __bf16* row = src + (size_t)m * ld;
  v16bf a;
#pragma unroll
  for (int j = 0; j < 16; ++j) a[j] = row[kh + j];
  return a;
}

// B fragment, 32x16 bf16 (KxN), built from row-major W[n][k] (i.e. B = W^T
// tile). Lanes 0-15 hold K=0..15 for column N=lane; lanes 16-31 hold K=16..31.
// Per-lane read is 16 contiguous bf16 -> vectorizes to b128 loads.
__device__ __forceinline__ v16bf load_bT_frag(const __bf16* __restrict__ w, int ldw) {
  const int lane = threadIdx.x & 31;
  const int n    = lane & 15;
  const int kb   = (lane >> 4) << 4;  // 0 or 16
  const __bf16* p = w + (size_t)n * ldw + kb;
  v16bf b;
#pragma unroll
  for (int j = 0; j < 16; ++j) b[j] = p[j];
  return b;
}

__device__ __forceinline__ v8f wmma_bf16(v16bf a, v16bf b, v8f c) {
  return __builtin_amdgcn_wmma_f32_16x16x32_bf16(false, a, false, b, (short)0, c,
                                                 false, false);
}

// One K segment (NSTEP * 32 wide) of a rank-4-output GEMM: a single A stream
// against four weight row-blocks, accumulating into four 16x16 fp32 tiles.
// Fully unrolled; all four B fragments are loaded before the wmmas so the
// scheduler can overlap next-iteration loads with current-matrix math.
template <int NSTEP>
__device__ __forceinline__ void k_block(
    const __bf16* __restrict__ asrc, int lda,
    const __bf16* __restrict__ w0, const __bf16* __restrict__ w1,
    const __bf16* __restrict__ w2, const __bf16* __restrict__ w3,
    int ldw, v8f& acc0, v8f& acc1, v8f& acc2, v8f& acc3) {
#pragma unroll
  for (int s = 0; s < NSTEP; ++s) {
    const v16bf a  = load_a_frag(asrc + 32 * s, lda);
    const v16bf b0 = load_bT_frag(w0 + 32 * s, ldw);
    const v16bf b1 = load_bT_frag(w1 + 32 * s, ldw);
    const v16bf b2 = load_bT_frag(w2 + 32 * s, ldw);
    const v16bf b3 = load_bT_frag(w3 + 32 * s, ldw);
    acc0 = wmma_bf16(a, b0, acc0);
    acc1 = wmma_bf16(a, b1, acc1);
    acc2 = wmma_bf16(a, b2, acc2);
    acc3 = wmma_bf16(a, b3, acc3);
  }
}

__device__ __forceinline__ float sigmoidf_(float x) {
  return 1.0f / (1.0f + __expf(-x));
}

// Device-scope software barrier for the persistent sequential kernel.
// Monotonic counter; caller supplies running target = n_syncs * NWG.
__device__ __forceinline__ void grid_sync(unsigned* ctr, unsigned target) {
  __builtin_amdgcn_fence(__ATOMIC_RELEASE, "agent");
  __syncthreads();
  if (threadIdx.x == 0) {
    atomicAdd(ctr, 1u);
    while (__hip_atomic_load(ctr, __ATOMIC_RELAXED, __HIP_MEMORY_SCOPE_AGENT) < target)
      __builtin_amdgcn_s_sleep(1);
  }
  __syncthreads();
  __builtin_amdgcn_fence(__ATOMIC_ACQUIRE, "agent");
}

// --------------------------- prep: weights -> bf16 -------------------------

__global__ __launch_bounds__(256) void prep_kernel(
    const float* __restrict__ x,
    const float* __restrict__ eWih1, const float* __restrict__ eWhh1,
    const float* __restrict__ ebih1, const float* __restrict__ ebhh1,
    const float* __restrict__ eWih2, const float* __restrict__ eWhh2,
    const float* __restrict__ ebih2, const float* __restrict__ ebhh2,
    const float* __restrict__ dWih1, const float* __restrict__ dWhh1,
    const float* __restrict__ dbih1, const float* __restrict__ dbhh1,
    const float* __restrict__ dWih2, const float* __restrict__ dWhh2,
    const float* __restrict__ dbih2, const float* __restrict__ dbhh2,
    const float* __restrict__ fc1w, const float* __restrict__ fc1b,
    const float* __restrict__ fc2w,
    __bf16* __restrict__ xbf,
    __bf16* __restrict__ Wce1, __bf16* __restrict__ Wce2,
    __bf16* __restrict__ Wcd1, __bf16* __restrict__ Wcd2,
    __bf16* __restrict__ fc1bf, __bf16* __restrict__ fc2bf,
    float* __restrict__ be1, float* __restrict__ be2,
    float* __restrict__ bd1, float* __restrict__ bd2,
    float* __restrict__ fb1, unsigned* __restrict__ ctrs) {
  const int tid = blockIdx.x * blockDim.x + threadIdx.x;
  const int gs  = gridDim.x * blockDim.x;

  // x: [B][T][OBS] fp32 -> [T][B][OBS] bf16 (contiguous rows per timestep)
  for (int i = tid; i < B_ * T_ * OBS_; i += gs) {
    const int k  = i % OBS_;
    const int bt = i / OBS_;
    const int tt = bt % T_;
    const int bb = bt / T_;
    xbf[((size_t)tt * B_ + bb) * OBS_ + k] = (__bf16)x[i];
  }
  // enc cell1 cat weights: [4H][OBS+HID]
  {
    const int K = OBS_ + HID_;
    for (int i = tid; i < 4 * HID_ * K; i += gs) {
      const int k = i % K, n = i / K;
      const float v = (k < OBS_) ? eWih1[(size_t)n * OBS_ + k]
                                 : eWhh1[(size_t)n * HID_ + (k - OBS_)];
      Wce1[i] = (__bf16)v;
    }
  }
  // enc cell2 cat weights: [4L][HID+LAT]
  {
    const int K = HID_ + LAT_;
    for (int i = tid; i < 4 * LAT_ * K; i += gs) {
      const int k = i % K, n = i / K;
      const float v = (k < HID_) ? eWih2[(size_t)n * HID_ + k]
                                 : eWhh2[(size_t)n * LAT_ + (k - HID_)];
      Wce2[i] = (__bf16)v;
    }
  }
  // dec cell1 cat weights: [4H][LAT+HID]
  {
    const int K = LAT_ + HID_;
    for (int i = tid; i < 4 * HID_ * K; i += gs) {
      const int k = i % K, n = i / K;
      const float v = (k < LAT_) ? dWih1[(size_t)n * LAT_ + k]
                                 : dWhh1[(size_t)n * HID_ + (k - LAT_)];
      Wcd1[i] = (__bf16)v;
    }
  }
  // dec cell2 cat weights: [4L][HID+LAT]
  {
    const int K = HID_ + LAT_;
    for (int i = tid; i < 4 * LAT_ * K; i += gs) {
      const int k = i % K, n = i / K;
      const float v = (k < HID_) ? dWih2[(size_t)n * HID_ + k]
                                 : dWhh2[(size_t)n * LAT_ + (k - HID_)];
      Wcd2[i] = (__bf16)v;
    }
  }
  for (int i = tid; i < HID_ * LAT_; i += gs) fc1bf[i] = (__bf16)fc1w[i];
  for (int i = tid; i < OBS_ * HID_; i += gs) fc2bf[i] = (__bf16)fc2w[i];
  for (int i = tid; i < 4 * HID_; i += gs) be1[i] = ebih1[i] + ebhh1[i];
  for (int i = tid; i < 4 * LAT_; i += gs) be2[i] = ebih2[i] + ebhh2[i];
  for (int i = tid; i < 4 * HID_; i += gs) bd1[i] = dbih1[i] + dbhh1[i];
  for (int i = tid; i < 4 * LAT_; i += gs) bd2[i] = dbih2[i] + dbhh2[i];
  for (int i = tid; i < HID_; i += gs) fb1[i] = fc1b[i];
  if (tid == 0) { ctrs[0] = 0u; ctrs[1] = 0u; }
}

// --------------------- persistent 2-layer LSTM sequence --------------------
// One launch runs all T steps for a (IN_DIM -> HID -> LAT) stack.
// h double-buffered bf16 in global; c fp32 in global but touched only by its
// owning wave (fixed tile assignment), so no extra sync needed for c.
// Templated on IN_DIM so every K trip count is a compile-time constant.

template <int IN_DIM>
__global__ __launch_bounds__(THREADS) void lstm_seq_kernel(
    const __bf16* __restrict__ in_seq,
    const __bf16* __restrict__ W1, const float* __restrict__ b1,  // [4H][IN_DIM+HID]
    const __bf16* __restrict__ W2, const float* __restrict__ b2,  // [4L][HID+LAT]
    __bf16* __restrict__ h1buf, __bf16* __restrict__ h2buf,
    float* __restrict__ c1, float* __restrict__ c2,
    __bf16* __restrict__ out_seq, unsigned* __restrict__ ctr) {
  constexpr int K1 = IN_DIM + HID_;
  constexpr int K2 = HID_ + LAT_;
  const int wid  = (blockIdx.x * blockDim.x + threadIdx.x) >> 5;
  const int lane = threadIdx.x & 31;
  const int H1SZ = B_ * HID_;
  const int H2SZ = B_ * LAT_;

  // Zero-initialize state (t=0 uses h=c=0), then barrier.
  {
    const int gtid = blockIdx.x * blockDim.x + threadIdx.x;
    const int nthr = NWG * THREADS;
    for (int i = gtid; i < 2 * H1SZ; i += nthr) h1buf[i] = (__bf16)0.f;
    for (int i = gtid; i < 2 * H2SZ; i += nthr) h2buf[i] = (__bf16)0.f;
    for (int i = gtid; i < H1SZ; i += nthr) c1[i] = 0.f;
    for (int i = gtid; i < H2SZ; i += nthr) c2[i] = 0.f;
  }
  unsigned nsync = 1;
  grid_sync(ctr, nsync * NWG);

  for (int t = 0; t < T_; ++t) {
    const int p = t & 1, pn = p ^ 1;

    // ---------------- cell 1: gates over [x_t | h1_prev], hidden = HID ----
    const __bf16* xrow = in_seq + (size_t)t * B_ * IN_DIM;
    const __bf16* h1p  = h1buf + (size_t)p * H1SZ;
    __bf16*       h1n  = h1buf + (size_t)pn * H1SZ;
    const int NT1 = (HID_ / 16) * (B_ / 16);  // 128 tile-pairs
    for (int tp = wid; tp < NT1; tp += TOTAL_WAVES) {
      const int mt = tp & 3, nt = tp >> 2;
      const int m0 = mt * 16, c0 = nt * 16;
      const int nl = c0 + (lane & 15);
      v8f ai, af, ag, ao;
      {
        const float bi0 = b1[0 * HID_ + nl];
        const float bf0 = b1[1 * HID_ + nl];
        const float bg0 = b1[2 * HID_ + nl];
        const float bo0 = b1[3 * HID_ + nl];
#pragma unroll
        for (int r = 0; r < 8; ++r) { ai[r] = bi0; af[r] = bf0; ag[r] = bg0; ao[r] = bo0; }
      }
      const __bf16* wi = W1 + ((size_t)(0 * HID_ + c0)) * K1;
      const __bf16* wf = W1 + ((size_t)(1 * HID_ + c0)) * K1;
      const __bf16* wg = W1 + ((size_t)(2 * HID_ + c0)) * K1;
      const __bf16* wo = W1 + ((size_t)(3 * HID_ + c0)) * K1;
      __builtin_prefetch(wi, 0, 1);
      // x segment (K = IN_DIM), then recurrent segment (K = HID).
      k_block<IN_DIM / 32>(xrow + (size_t)m0 * IN_DIM, IN_DIM,
                           wi, wf, wg, wo, K1, ai, af, ag, ao);
      k_block<HID_ / 32>(h1p + (size_t)m0 * HID_, HID_,
                         wi + IN_DIM, wf + IN_DIM, wg + IN_DIM, wo + IN_DIM,
                         K1, ai, af, ag, ao);
      const int mbase = m0 + ((lane >> 4) << 3);
#pragma unroll
      for (int r = 0; r < 8; ++r) {
        const int m = mbase + r;
        const float iv = sigmoidf_(ai[r]);
        const float fv = sigmoidf_(af[r]);
        const float gv = tanhf(ag[r]);
        const float ov = sigmoidf_(ao[r]);
        const size_t ci = (size_t)m * HID_ + nl;
        const float cn = fv * c1[ci] + iv * gv;
        c1[ci] = cn;
        h1n[ci] = (__bf16)(ov * tanhf(cn));
      }
    }
    grid_sync(ctr, (++nsync) * NWG);

    // ---------------- cell 2: gates over [h1_new | h2_prev], hidden = LAT -
    const __bf16* h2p = h2buf + (size_t)p * H2SZ;
    __bf16*       h2n = h2buf + (size_t)pn * H2SZ;
    __bf16*       orow = out_seq + (size_t)t * B_ * LAT_;
    const int NT2 = (LAT_ / 16) * (B_ / 16);  // 64 tile-pairs
    for (int tp = wid; tp < NT2; tp += TOTAL_WAVES) {
      const int mt = tp & 3, nt = tp >> 2;
      const int m0 = mt * 16, c0 = nt * 16;
      const int nl = c0 + (lane & 15);
      v8f ai, af, ag, ao;
      {
        const float bi0 = b2[0 * LAT_ + nl];
        const float bf0 = b2[1 * LAT_ + nl];
        const float bg0 = b2[2 * LAT_ + nl];
        const float bo0 = b2[3 * LAT_ + nl];
#pragma unroll
        for (int r = 0; r < 8; ++r) { ai[r] = bi0; af[r] = bf0; ag[r] = bg0; ao[r] = bo0; }
      }
      const __bf16* wi = W2 + ((size_t)(0 * LAT_ + c0)) * K2;
      const __bf16* wf = W2 + ((size_t)(1 * LAT_ + c0)) * K2;
      const __bf16* wg = W2 + ((size_t)(2 * LAT_ + c0)) * K2;
      const __bf16* wo = W2 + ((size_t)(3 * LAT_ + c0)) * K2;
      __builtin_prefetch(wi, 0, 1);
      // h1_new segment (K = HID), then recurrent segment (K = LAT).
      k_block<HID_ / 32>(h1n + (size_t)m0 * HID_, HID_,
                         wi, wf, wg, wo, K2, ai, af, ag, ao);
      k_block<LAT_ / 32>(h2p + (size_t)m0 * LAT_, LAT_,
                         wi + HID_, wf + HID_, wg + HID_, wo + HID_,
                         K2, ai, af, ag, ao);
      const int mbase = m0 + ((lane >> 4) << 3);
#pragma unroll
      for (int r = 0; r < 8; ++r) {
        const int m = mbase + r;
        const float iv = sigmoidf_(ai[r]);
        const float fv = sigmoidf_(af[r]);
        const float gv = tanhf(ag[r]);
        const float ov = sigmoidf_(ao[r]);
        const size_t ci = (size_t)m * LAT_ + nl;
        const float cn = fv * c2[ci] + iv * gv;
        c2[ci] = cn;
        const __bf16 hv = (__bf16)(ov * tanhf(cn));
        h2n[ci]  = hv;
        orow[ci] = hv;
      }
    }
    grid_sync(ctr, (++nsync) * NWG);
  }
}

// ------------------------- fused FC head (relu + proj) ---------------------
// Per block: 64 rows of d (flat r = t*B + b). GEMM1 (LAT->HID)+bias+ReLU into
// LDS bf16, then GEMM2 (HID->OBS) to fp32 output in [B][T][OBS] order.
// Each wave computes FOUR 16-wide N tiles sharing one A stream (same shape as
// the gate GEMM -> same pipelined k_block codegen, 4x A reuse).

__global__ __launch_bounds__(256) void fc_fused_kernel(
    const __bf16* __restrict__ dseq,            // [T*B][LAT]
    const __bf16* __restrict__ w1,              // [HID][LAT]
    const float* __restrict__ b1f,              // [HID]
    const __bf16* __restrict__ w2,              // [OBS][HID]
    float* __restrict__ out) {                  // [B][T][OBS]
  __shared__ __bf16 hbuf[B_ * HID_];            // 64 KB
  const int r0   = blockIdx.x * 64;
  const int wid  = threadIdx.x >> 5;
  const int lane = threadIdx.x & 31;

  // GEMM1 + ReLU -> LDS.  Tile-groups: 4 m-tiles x (HID/64)=8 n-groups = 32.
  for (int tp = wid; tp < 4 * (HID_ / 64); tp += 8) {
    const int mt = tp & 3, ng = tp >> 2;
    const int m0 = mt * 16, c0 = ng * 64;
    v8f a0, a1, a2, a3;
    {
      const float bv0 = b1f[c0 + 0  + (lane & 15)];
      const float bv1 = b1f[c0 + 16 + (lane & 15)];
      const float bv2 = b1f[c0 + 32 + (lane & 15)];
      const float bv3 = b1f[c0 + 48 + (lane & 15)];
#pragma unroll
      for (int r = 0; r < 8; ++r) { a0[r] = bv0; a1[r] = bv1; a2[r] = bv2; a3[r] = bv3; }
    }
    k_block<LAT_ / 32>(dseq + (size_t)(r0 + m0) * LAT_, LAT_,
                       w1 + (size_t)(c0 + 0)  * LAT_, w1 + (size_t)(c0 + 16) * LAT_,
                       w1 + (size_t)(c0 + 32) * LAT_, w1 + (size_t)(c0 + 48) * LAT_,
                       LAT_, a0, a1, a2, a3);
    const int mbase = m0 + ((lane >> 4) << 3);
    const int nl = lane & 15;
#pragma unroll
    for (int r = 0; r < 8; ++r) {
      hbuf[(mbase + r) * HID_ + c0 + 0  + nl] = (__bf16)fmaxf(a0[r], 0.f);
      hbuf[(mbase + r) * HID_ + c0 + 16 + nl] = (__bf16)fmaxf(a1[r], 0.f);
      hbuf[(mbase + r) * HID_ + c0 + 32 + nl] = (__bf16)fmaxf(a2[r], 0.f);
      hbuf[(mbase + r) * HID_ + c0 + 48 + nl] = (__bf16)fmaxf(a3[r], 0.f);
    }
  }
  __syncthreads();

  // GEMM2 from LDS -> output.  4 m-tiles x (OBS/64)=2 n-groups = 8 groups.
  for (int tp = wid; tp < 4 * (OBS_ / 64); tp += 8) {
    const int mt = tp & 3, ng = tp >> 2;
    const int m0 = mt * 16, c0 = ng * 64;
    v8f a0 = {}, a1 = {}, a2 = {}, a3 = {};
    k_block<HID_ / 32>(&hbuf[m0 * HID_], HID_,
                       w2 + (size_t)(c0 + 0)  * HID_, w2 + (size_t)(c0 + 16) * HID_,
                       w2 + (size_t)(c0 + 32) * HID_, w2 + (size_t)(c0 + 48) * HID_,
                       HID_, a0, a1, a2, a3);
    const int mbase = m0 + ((lane >> 4) << 3);
    const int nl = lane & 15;
#pragma unroll
    for (int r = 0; r < 8; ++r) {
      const int rr = r0 + mbase + r;
      const int tt = rr / B_;
      const int bb = rr % B_;
      float* orow = out + ((size_t)bb * T_ + tt) * OBS_;
      orow[c0 + 0  + nl] = a0[r];
      orow[c0 + 16 + nl] = a1[r];
      orow[c0 + 32 + nl] = a2[r];
      orow[c0 + 48 + nl] = a3[r];
    }
  }
}

// -------------------------------- host side --------------------------------

extern "C" void kernel_launch(void* const* d_in, const int* in_sizes, int n_in,
                              void* d_out, int out_size, void* d_ws, size_t ws_size,
                              hipStream_t stream) {
  (void)in_sizes; (void)n_in; (void)out_size; (void)ws_size;

  const float* x     = (const float*)d_in[0];
  const float* eWih1 = (const float*)d_in[1];
  const float* eWhh1 = (const float*)d_in[2];
  const float* ebih1 = (const float*)d_in[3];
  const float* ebhh1 = (const float*)d_in[4];
  const float* eWih2 = (const float*)d_in[5];
  const float* eWhh2 = (const float*)d_in[6];
  const float* ebih2 = (const float*)d_in[7];
  const float* ebhh2 = (const float*)d_in[8];
  const float* dWih1 = (const float*)d_in[9];
  const float* dWhh1 = (const float*)d_in[10];
  const float* dbih1 = (const float*)d_in[11];
  const float* dbhh1 = (const float*)d_in[12];
  const float* dWih2 = (const float*)d_in[13];
  const float* dWhh2 = (const float*)d_in[14];
  const float* dbih2 = (const float*)d_in[15];
  const float* dbhh2 = (const float*)d_in[16];
  const float* fc1w  = (const float*)d_in[17];
  const float* fc1b  = (const float*)d_in[18];
  const float* fc2w  = (const float*)d_in[19];
  float* out = (float*)d_out;

  char* ws = (char*)d_ws;
  size_t off = 0;
  auto alloc = [&](size_t bytes) -> void* {
    void* p = ws + off;
    off = (off + bytes + 255) & ~(size_t)255;
    return p;
  };

  __bf16* xbf  = (__bf16*)alloc((size_t)T_ * B_ * OBS_ * 2);
  __bf16* zbf  = (__bf16*)alloc((size_t)T_ * B_ * LAT_ * 2);
  __bf16* dbf  = (__bf16*)alloc((size_t)T_ * B_ * LAT_ * 2);
  __bf16* Wce1 = (__bf16*)alloc((size_t)4 * HID_ * (OBS_ + HID_) * 2);
  __bf16* Wce2 = (__bf16*)alloc((size_t)4 * LAT_ * (HID_ + LAT_) * 2);
  __bf16* Wcd1 = (__bf16*)alloc((size_t)4 * HID_ * (LAT_ + HID_) * 2);
  __bf16* Wcd2 = (__bf16*)alloc((size_t)4 * LAT_ * (HID_ + LAT_) * 2);
  __bf16* fc1bf = (__bf16*)alloc((size_t)HID_ * LAT_ * 2);
  __bf16* fc2bf = (__bf16*)alloc((size_t)OBS_ * HID_ * 2);
  float* be1 = (float*)alloc(4 * HID_ * 4);
  float* be2 = (float*)alloc(4 * LAT_ * 4);
  float* bd1 = (float*)alloc(4 * HID_ * 4);
  float* bd2 = (float*)alloc(4 * LAT_ * 4);
  float* fb1 = (float*)alloc(HID_ * 4);
  __bf16* h1buf = (__bf16*)alloc((size_t)2 * B_ * HID_ * 2);
  __bf16* h2buf = (__bf16*)alloc((size_t)2 * B_ * LAT_ * 2);
  float* c1 = (float*)alloc((size_t)B_ * HID_ * 4);
  float* c2 = (float*)alloc((size_t)B_ * LAT_ * 4);
  unsigned* ctrs = (unsigned*)alloc(256);

  prep_kernel<<<512, 256, 0, stream>>>(
      x, eWih1, eWhh1, ebih1, ebhh1, eWih2, eWhh2, ebih2, ebhh2,
      dWih1, dWhh1, dbih1, dbhh1, dWih2, dWhh2, dbih2, dbhh2,
      fc1w, fc1b, fc2w,
      xbf, Wce1, Wce2, Wcd1, Wcd2, fc1bf, fc2bf,
      be1, be2, bd1, bd2, fb1, ctrs);

  // Encoder: x (OBS) -> h1 (HID) -> z (LAT)
  lstm_seq_kernel<OBS_><<<NWG, THREADS, 0, stream>>>(
      xbf, Wce1, be1, Wce2, be2, h1buf, h2buf, c1, c2, zbf, &ctrs[0]);

  // Decoder: z (LAT) -> h1 (HID) -> d (LAT)
  lstm_seq_kernel<LAT_><<<NWG, THREADS, 0, stream>>>(
      zbf, Wcd1, bd1, Wcd2, bd2, h1buf, h2buf, c1, c2, dbf, &ctrs[1]);

  // FC head: out = relu(d @ fc1^T + b) @ fc2^T
  fc_fused_kernel<<<(T_ * B_) / 64, 256, 0, stream>>>(dbf, fc1bf, fb1, fc2bf, out);
}
